// ViT22BUnet_27779848470592
// MI455X (gfx1250) — matrix operand order repaired and checked by reading, not compiled, not run on hardware
//
#include <hip/hip_runtime.h>
#include <math.h>

// ---------------------------------------------------------------------------
// Types
// ---------------------------------------------------------------------------
typedef __attribute__((ext_vector_type(16))) _Float16 v16h;
typedef __attribute__((ext_vector_type(8)))  _Float16 v8h;
typedef __attribute__((ext_vector_type(8)))  float    v8f;

#define LDSS 40   // LDS row stride in halves (32 data + 8 pad)

__device__ inline v8h zero8h() {
  v8h z;
#pragma unroll
  for (int e = 0; e < 8; ++e) z[e] = (_Float16)0.f;
  return z;
}
__device__ inline v8f zero8f() {
  v8f z;
#pragma unroll
  for (int e = 0; e < 8; ++e) z[e] = 0.f;
  return z;
}

// ---------------------------------------------------------------------------
// Batched WMMA GEMM:  C[z] = act( alpha * A[z] * B[z]^T' + bias + res )
//   A: f16 [M,K] row-major (lda), batch offset zb*sAb + zh*sAh, z = zb*H+zh
//   B: f16, bLayout==0 -> [N,K] row-major (weights / K-matrix)
//           bLayout==1 -> [K,N] row-major (V-matrix)
//   C: f32 [M,N] (ldc)
// ---------------------------------------------------------------------------
struct GemmP {
  const _Float16* A; const _Float16* B; float* C;
  const float* bias; const float* res;
  long lda, ldb, ldc;
  long sAb, sAh, sBb, sBh, sCb, sCh, sRb, sRh;
  int M, N, K, H, bLayout, act;   // act: 0 none, 1 exact GELU
  float alpha;
};

__global__ void __launch_bounds__(256) k_gemm_wmma(GemmP p) {
  __shared__ _Float16 As[128 * LDSS];
  __shared__ _Float16 Bs[128 * LDSS];

  const int tid  = threadIdx.x;
  const int lane = tid & 31;
  const int wave = tid >> 5;
  const int z  = blockIdx.z;
  const int zb = z / p.H, zh = z % p.H;

  const _Float16* A  = p.A + (long)zb * p.sAb + (long)zh * p.sAh;
  const _Float16* Bm = p.B + (long)zb * p.sBb + (long)zh * p.sBh;
  float* C = p.C + (long)zb * p.sCb + (long)zh * p.sCh;
  const float* res = p.res ? (p.res + (long)zb * p.sRb + (long)zh * p.sRh) : nullptr;

  const int row0 = blockIdx.y * 128;
  const int col0 = blockIdx.x * 128;

  const int wm = wave >> 1;   // 0..3 -> M sub-band of 32
  const int wn = wave & 1;    // 0..1 -> N sub-band of 64
  const int lh = lane >> 4;   // lane half
  const int lr = lane & 15;

  v8f acc[2][4];
#pragma unroll
  for (int i = 0; i < 2; ++i)
#pragma unroll
    for (int j = 0; j < 4; ++j) acc[i][j] = zero8f();

  for (int k0 = 0; k0 < p.K; k0 += 32) {
    // ---- cooperative LDS fill: A tile (128 x 32 halves) ----
#pragma unroll
    for (int s = 0; s < 2; ++s) {
      int seg = tid * 2 + s;            // 512 segments of 8 halves
      int r = seg >> 2;
      int c = (seg & 3) * 8;
      v8h val = zero8h();
      long gr = (long)row0 + r;
      if (gr < p.M && (k0 + c) < p.K)
        val = *(const v8h*)(A + gr * p.lda + k0 + c);
      *(v8h*)(&As[r * LDSS + c]) = val;
    }
    // ---- B tile into Bs[n][k] ----
    if (p.bLayout == 0) {
#pragma unroll
      for (int s = 0; s < 2; ++s) {
        int seg = tid * 2 + s;
        int r = seg >> 2;               // n
        int c = (seg & 3) * 8;          // k
        v8h val = zero8h();
        long gn = (long)col0 + r;
        if (gn < p.N && (k0 + c) < p.K)
          val = *(const v8h*)(Bm + gn * p.ldb + k0 + c);
        *(v8h*)(&Bs[r * LDSS + c]) = val;
      }
    } else {
#pragma unroll
      for (int e = 0; e < 16; ++e) {
        int idx = tid * 16 + e;         // 4096 elements
        int r = idx >> 5;               // n
        int c = idx & 31;               // k
        _Float16 val = (_Float16)0.f;
        long gn = (long)col0 + r;
        if (gn < p.N && (k0 + c) < p.K)
          val = Bm[(long)(k0 + c) * p.ldb + gn];
        Bs[r * LDSS + c] = val;
      }
    }
    __syncthreads();

    // prefetch next K tile (gfx1250 global_prefetch_b8)
    if (k0 + 32 < p.K) {
      long pr = (long)row0 + (tid & 127);
      if (pr < p.M) __builtin_prefetch(A + pr * p.lda + k0 + 32, 0, 1);
      if (p.bLayout == 0) {
        long pn = (long)col0 + (tid & 127);
        if (pn < p.N) __builtin_prefetch(Bm + pn * p.ldb + k0 + 32, 0, 1);
      }
    }

#pragma unroll
    for (int mi = 0; mi < 2; ++mi) {
      const int ar = wm * 32 + mi * 16 + lr;
      // A fragment: lane-half 0 holds K {0..7,16..23}, half 1 holds {8..15,24..31}
      v8h alo = *(const v8h*)(&As[ar * LDSS + lh * 8]);
      v8h ahi = *(const v8h*)(&As[ar * LDSS + lh * 8 + 16]);
      v16h afrag;
#pragma unroll
      for (int e = 0; e < 8; ++e) { afrag[e] = alo[e]; afrag[e + 8] = ahi[e]; }
#pragma unroll
      for (int ni = 0; ni < 4; ++ni) {
        const int bn = wn * 64 + ni * 16 + lr;
        // B fragment: lane-half 0 holds K {0..7,16..23}, half 1 holds {8..15,24..31}
        v8h blo = *(const v8h*)(&Bs[bn * LDSS + lh * 8]);
        v8h bhi = *(const v8h*)(&Bs[bn * LDSS + lh * 8 + 16]);
        v16h bfrag;
#pragma unroll
        for (int e = 0; e < 8; ++e) { bfrag[e] = blo[e]; bfrag[e + 8] = bhi[e]; }
        acc[mi][ni] = __builtin_amdgcn_wmma_f32_16x16x32_f16(
            false, afrag, false, bfrag, (short)0, acc[mi][ni], false, false);
      }
    }
    __syncthreads();
  }

  // ---- epilogue: C/D layout -> lane l, elem v: m = (l>>4)*8+v, n = l&15 ----
#pragma unroll
  for (int mi = 0; mi < 2; ++mi) {
#pragma unroll
    for (int ni = 0; ni < 4; ++ni) {
      const int tr0 = row0 + wm * 32 + mi * 16 + lh * 8;
      const int tc  = col0 + wn * 64 + ni * 16 + lr;
      if (tc >= p.N) continue;
#pragma unroll
      for (int v = 0; v < 8; ++v) {
        int gm = tr0 + v;
        if (gm >= p.M) continue;
        float val = acc[mi][ni][v] * p.alpha;
        if (p.bias) val += p.bias[tc];
        if (res)    val += res[(long)gm * p.ldc + tc];
        if (p.act == 1) val = 0.5f * val * (1.f + erff(val * 0.70710678f));
        C[(long)gm * p.ldc + tc] = val;
      }
    }
  }
}

// ---------------------------------------------------------------------------
// Elementwise / reduction kernels
// ---------------------------------------------------------------------------
__global__ void k_pack_f16(const float* __restrict__ src, _Float16* __restrict__ dst, long n) {
  long stride = (long)gridDim.x * blockDim.x;
  for (long i = (long)blockIdx.x * blockDim.x + threadIdx.x; i < n; i += stride)
    dst[i] = (_Float16)src[i];
}

// LayerNorm (no affine). If mod>0: row normalized only if (row % mod) < lim.
// out = ln(x) + (res ? res : 0). Safe for out==x and out==res.
__global__ void __launch_bounds__(256) k_layernorm(const float* __restrict__ x,
                                                   float* __restrict__ out,
                                                   const float* __restrict__ res,
                                                   int len, int mod, int lim) {
  long row = blockIdx.x;
  if (mod) { int c = (int)(row % mod); if (c >= lim) return; }
  const float* xr = x + row * (long)len;
  float* orow = out + row * (long)len;
  __shared__ float sbuf[256];
  float s = 0.f;
  for (int j = threadIdx.x; j < len; j += 256) s += xr[j];
  sbuf[threadIdx.x] = s; __syncthreads();
  for (int st = 128; st; st >>= 1) {
    if (threadIdx.x < st) sbuf[threadIdx.x] += sbuf[threadIdx.x + st];
    __syncthreads();
  }
  float mean = sbuf[0] / len; __syncthreads();
  float vs = 0.f;
  for (int j = threadIdx.x; j < len; j += 256) { float d = xr[j] - mean; vs += d * d; }
  sbuf[threadIdx.x] = vs; __syncthreads();
  for (int st = 128; st; st >>= 1) {
    if (threadIdx.x < st) sbuf[threadIdx.x] += sbuf[threadIdx.x + st];
    __syncthreads();
  }
  float inv = rsqrtf(sbuf[0] / len + 1e-5f);
  for (int j = threadIdx.x; j < len; j += 256) {
    float v = (xr[j] - mean) * inv;
    if (res) v += res[row * (long)len + j];
    orow[j] = v;
  }
}

__global__ void __launch_bounds__(256) k_softmax(float* __restrict__ x, int len) {
  long row = blockIdx.x;
  float* xr = x + row * (long)len;
  __shared__ float sbuf[256];
  float m = -3.4e38f;
  for (int j = threadIdx.x; j < len; j += 256) m = fmaxf(m, xr[j]);
  sbuf[threadIdx.x] = m; __syncthreads();
  for (int st = 128; st; st >>= 1) {
    if (threadIdx.x < st) sbuf[threadIdx.x] = fmaxf(sbuf[threadIdx.x], sbuf[threadIdx.x + st]);
    __syncthreads();
  }
  m = sbuf[0]; __syncthreads();
  float s = 0.f;
  for (int j = threadIdx.x; j < len; j += 256) { float e = __expf(xr[j] - m); xr[j] = e; s += e; }
  sbuf[threadIdx.x] = s; __syncthreads();
  for (int st = 128; st; st >>= 1) {
    if (threadIdx.x < st) sbuf[threadIdx.x] += sbuf[threadIdx.x + st];
    __syncthreads();
  }
  float inv = 1.f / sbuf[0];
  for (int j = threadIdx.x; j < len; j += 256) xr[j] *= inv;
}

// dst[b,n,j] = src[b,n,j*df] + pos[n*1024 + j*step]   (df==1 allows src==dst)
__global__ void k_down_add_pos(const float* __restrict__ src, float* __restrict__ dst,
                               const float* __restrict__ pos, int n, int Dnew, int df, int step) {
  long total = 2L * n * Dnew;
  long stride = (long)gridDim.x * blockDim.x;
  for (long i = (long)blockIdx.x * blockDim.x + threadIdx.x; i < total; i += stride) {
    long row = i / Dnew; int j = (int)(i % Dnew); int nn = (int)(row % n);
    dst[i] = src[row * (long)Dnew * df + (long)j * df] + pos[(long)nn * 1024 + (long)j * step];
  }
}

__global__ void k_copy_rows(const float* __restrict__ src, float* __restrict__ dst,
                            int rows, int D, long sSrcB, long sDstB, int rowOff) {
  long total = 2L * rows * D;
  long stride = (long)gridDim.x * blockDim.x;
  for (long i = (long)blockIdx.x * blockDim.x + threadIdx.x; i < total; i += stride) {
    int b = (int)(i / ((long)rows * D));
    int r = (int)((i / D) % rows);
    int j = (int)(i % D);
    dst[b * sDstB + (long)(rowOff + r) * D + j] = src[b * sSrcB + (long)r * D + j];
  }
}

// feature upsample x2 into concat destination rows
__global__ void k_up_rows(const float* __restrict__ src, float* __restrict__ dst,
                          int rows, int Dsrc, long sSrcB, long sDstB, int rowOff) {
  long total = 2L * rows * Dsrc;
  long stride = (long)gridDim.x * blockDim.x;
  for (long i = (long)blockIdx.x * blockDim.x + threadIdx.x; i < total; i += stride) {
    int b = (int)(i / ((long)rows * Dsrc));
    int r = (int)((i / Dsrc) % rows);
    int j = (int)(i % Dsrc);
    float v = src[b * sSrcB + (long)r * Dsrc + j];
    float* d = dst + b * sDstB + (long)(rowOff + r) * (2 * Dsrc) + 2 * j;
    d[0] = v; d[1] = v;
  }
}

// x[2,3,256,256] -> xp f16 [2,256,768]  (b c (h p1) (w p2) -> b (h w) (p1 p2 c))
__global__ void k_patchify(const float* __restrict__ x, _Float16* __restrict__ xp) {
  long total = 2L * 256 * 768;
  long stride = (long)gridDim.x * blockDim.x;
  for (long i = (long)blockIdx.x * blockDim.x + threadIdx.x; i < total; i += stride) {
    int pd = (int)(i % 768);
    int nn = (int)((i / 768) % 256);
    int b  = (int)(i / (768L * 256));
    int p1 = pd / 48, rem = pd % 48, p2 = rem / 3, c = rem % 3;
    int hh = nn >> 4, ww = nn & 15;
    xp[i] = (_Float16)x[(((long)b * 3 + c) * 256 + hh * 16 + p1) * 256 + ww * 16 + p2];
  }
}

// t [2,1024,768] -> y [2,12,256,256]
__global__ void k_unpatchify(const float* __restrict__ t, float* __restrict__ y) {
  long total = 2L * 12 * 65536;
  long stride = (long)gridDim.x * blockDim.x;
  for (long i = (long)blockIdx.x * blockDim.x + threadIdx.x; i < total; i += stride) {
    int xx = (int)(i & 255);
    int yy = (int)((i >> 8) & 255);
    int ch = (int)((i >> 16) % 12);
    int b  = (int)(i / (12L * 65536));
    int tt = ch / 3, c = ch % 3;
    int hh = yy >> 4, p1 = yy & 15, ww = xx >> 4, p2 = xx & 15;
    long row  = (long)tt * 256 + hh * 16 + ww;
    long feat = ((long)p1 * 16 + p2) * 3 + c;
    y[i] = t[((long)b * 1024 + row) * 768 + feat];
  }
}

// direct 3x3 conv with reflection pad 1; act: 0 none, 2 tanh
__global__ void __launch_bounds__(256) k_conv3x3(const float* __restrict__ in,
                                                 const float* __restrict__ w,
                                                 const float* __restrict__ bias,
                                                 float* __restrict__ out,
                                                 int IC, int OC, int act) {
  long total = 2L * OC * 65536;
  long stride = (long)gridDim.x * blockDim.x;
  for (long i = (long)blockIdx.x * blockDim.x + threadIdx.x; i < total; i += stride) {
    int xx = (int)(i & 255);
    int yy = (int)((i >> 8) & 255);
    int oc = (int)((i >> 16) % OC);
    int b  = (int)(i / ((long)OC << 16));
    float acc = bias[oc];
    for (int ic = 0; ic < IC; ++ic) {
      const float* ip = in + (((long)b * IC + ic) << 16);
      const float* wp = w + (long)(oc * IC + ic) * 9;
#pragma unroll
      for (int ky = 0; ky < 3; ++ky) {
        int iy = yy + ky - 1; iy = iy < 0 ? 1 : (iy > 255 ? 254 : iy);
#pragma unroll
        for (int kx = 0; kx < 3; ++kx) {
          int ix = xx + kx - 1; ix = ix < 0 ? 1 : (ix > 255 ? 254 : ix);
          acc += ip[(iy << 8) + ix] * wp[ky * 3 + kx];
        }
      }
    }
    if (act == 2) acc = tanhf(acc);
    out[i] = acc;
  }
}

// InstanceNorm over 256*256 per (b,c); out = (res?res:0) + inorm(x); optional relu
__global__ void __launch_bounds__(256) k_inorm(const float* __restrict__ x,
                                               float* __restrict__ out,
                                               const float* __restrict__ res, int relu) {
  long ch = blockIdx.x;
  const float* xr = x + ch * 65536L;
  float* orow = out + ch * 65536L;
  const float* rr = res ? res + ch * 65536L : nullptr;
  __shared__ float s1[256], s2[256];
  float a = 0.f, b2 = 0.f;
  for (int j = threadIdx.x; j < 65536; j += 256) { float v = xr[j]; a += v; b2 += v * v; }
  s1[threadIdx.x] = a; s2[threadIdx.x] = b2; __syncthreads();
  for (int st = 128; st; st >>= 1) {
    if (threadIdx.x < st) { s1[threadIdx.x] += s1[threadIdx.x + st]; s2[threadIdx.x] += s2[threadIdx.x + st]; }
    __syncthreads();
  }
  float mean = s1[0] * (1.f / 65536.f);
  float var  = s2[0] * (1.f / 65536.f) - mean * mean;
  float inv  = rsqrtf(var + 1e-5f);
  for (int j = threadIdx.x; j < 65536; j += 256) {
    float v = (xr[j] - mean) * inv;
    if (rr) v += rr[j];
    if (relu) v = fmaxf(v, 0.f);
    orow[j] = v;
  }
}

// ---------------------------------------------------------------------------
// Host orchestration
// ---------------------------------------------------------------------------
struct Bump {
  char* base; size_t off;
  void* get(size_t bytes) {
    size_t a = (off + 255) & ~(size_t)255;
    off = a + bytes;
    return base + a;
  }
};

struct LayerW {
  const float *wqkv, *wo, *bo, *w1, *b1, *w2, *b2;
  _Float16 *wqkvh, *woh, *w1h, *w2h;
  int d;
};

static inline int gblocks(long total) {
  long b = (total + 255) / 256;
  return (int)(b > 65535 ? 65535 : (b < 1 ? 1 : b));
}

extern "C" void kernel_launch(void* const* d_in, const int* in_sizes, int n_in,
                              void* d_out, int out_size, void* d_ws, size_t ws_size,
                              hipStream_t stream) {
  (void)in_sizes; (void)out_size; (void)ws_size;
  if (n_in < 67) return;

  const float* xin   = (const float*)d_in[0];
  const float* pos   = (const float*)d_in[1];
  const float* emb_w = (const float*)d_in[2];
  const float* emb_b = (const float*)d_in[3];

  LayerW L[7];
  const int base[7] = {4, 11, 18, 25, 32, 39, 46};
  const int dims[7] = {1024, 512, 256, 128, 256, 512, 1024};
  for (int i = 0; i < 7; ++i) {
    int b0 = base[i];
    L[i].wqkv = (const float*)d_in[b0 + 0];
    L[i].wo   = (const float*)d_in[b0 + 1];
    L[i].bo   = (const float*)d_in[b0 + 2];
    L[i].w1   = (const float*)d_in[b0 + 3];
    L[i].b1   = (const float*)d_in[b0 + 4];
    L[i].w2   = (const float*)d_in[b0 + 5];
    L[i].b2   = (const float*)d_in[b0 + 6];
    L[i].d = dims[i];
  }
  const float* out_w = (const float*)d_in[53];
  const float* out_b = (const float*)d_in[54];
  const float* cw1 = (const float*)d_in[55]; const float* cb1 = (const float*)d_in[56];
  const float* cw2 = (const float*)d_in[57]; const float* cb2 = (const float*)d_in[58];
  const float* rw1 = (const float*)d_in[59]; const float* rb1 = (const float*)d_in[60];
  const float* rw2 = (const float*)d_in[61]; const float* rb2 = (const float*)d_in[62];
  const float* cw3 = (const float*)d_in[63]; const float* cb3 = (const float*)d_in[64];
  const float* cw4 = (const float*)d_in[65]; const float* cb4 = (const float*)d_in[66];

  // ---------------- workspace layout ----------------
  Bump bump{(char*)d_ws, 0};
  _Float16* embw_h = (_Float16*)bump.get((size_t)1024 * 768 * 2);
  _Float16* outw_h = (_Float16*)bump.get((size_t)768 * 1024 * 2);
  for (int i = 0; i < 7; ++i) {
    size_t d = (size_t)dims[i];
    L[i].wqkvh = (_Float16*)bump.get(3 * d * d * 2);
    L[i].woh   = (_Float16*)bump.get(d * d * 2);
    L[i].w1h   = (_Float16*)bump.get(4 * d * d * 2);
    L[i].w2h   = (_Float16*)bump.get(4 * d * d * 2);
  }
  float* feat0 = (float*)bump.get((size_t)512 * 1024 * 4);
  float* feat1 = (float*)bump.get((size_t)512 * 512 * 4);
  float* feat2 = (float*)bump.get((size_t)512 * 256 * 4);
  float* bufX0 = (float*)bump.get((size_t)2048 * 1024 * 4);
  float* bufX1 = (float*)bump.get((size_t)2048 * 1024 * 4);
  float* y12   = (float*)bump.get((size_t)2 * 12 * 65536 * 4);
  size_t mark = bump.off;

  // phase A: transformer scratch
  _Float16* xh      = (_Float16*)bump.get((size_t)2048 * 1024 * 2);
  float*    qkvf    = (float*)bump.get((size_t)2048 * 3072 * 4);
  _Float16* qkvh    = (_Float16*)bump.get((size_t)2048 * 3072 * 2);
  float*    scoresf = (float*)bump.get((size_t)16 * 1024 * 1024 * 4);
  _Float16* scoresh = (_Float16*)bump.get((size_t)16 * 1024 * 1024 * 2);
  float*    avf     = (float*)bump.get((size_t)2048 * 1024 * 4);
  _Float16* avh     = (_Float16*)bump.get((size_t)2048 * 1024 * 2);
  float*    h1f     = (float*)bump.get((size_t)2048 * 4096 * 4);
  _Float16* h1h     = (_Float16*)bump.get((size_t)2048 * 4096 * 2);
  float*    parf    = (float*)bump.get((size_t)2048 * 1024 * 4);
  float*    sumf    = (float*)bump.get((size_t)2048 * 1024 * 4);

  // phase B: CNN scratch (overlays phase A; transformer fully drained first)
  bump.off = mark;
  float* c64   = (float*)bump.get((size_t)2 * 64 * 65536 * 4);
  float* c128A = (float*)bump.get((size_t)2 * 128 * 65536 * 4);
  float* c128B = (float*)bump.get((size_t)2 * 128 * 65536 * 4);
  float* c128C = (float*)bump.get((size_t)2 * 128 * 65536 * 4);

  // ---------------- helpers ----------------
  auto pack = [&](const float* s, _Float16* dst, long n) {
    k_pack_f16<<<gblocks(n), 256, 0, stream>>>(s, dst, n);
  };
  auto gemm = [&](const _Float16* A, long lda, const _Float16* B, long ldb, int bLayout,
                  float* C, long ldc, int M, int N, int K,
                  const float* bias, const float* res, float alpha, int act,
                  int Z, int H, long sAb, long sAh, long sBb, long sBh,
                  long sCb, long sCh) {
    GemmP p;
    p.A = A; p.B = B; p.C = C; p.bias = bias; p.res = res;
    p.lda = lda; p.ldb = ldb; p.ldc = ldc;
    p.sAb = sAb; p.sAh = sAh; p.sBb = sBb; p.sBh = sBh;
    p.sCb = sCb; p.sCh = sCh; p.sRb = 0; p.sRh = 0;
    p.M = M; p.N = N; p.K = K; p.H = H; p.bLayout = bLayout; p.act = act; p.alpha = alpha;
    dim3 grid((N + 127) / 128, (M + 127) / 128, Z);
    k_gemm_wmma<<<grid, 256, 0, stream>>>(p);
  };

  auto run_layer = [&](const LayerW& Lw, float* xb, int n) {
    const int d = Lw.d, m = 4 * d, dh = d / 8, rows = 2 * n;
    pack(xb, xh, (long)rows * d);
    // qkv = x @ Wqkv^T  (no bias in reference)
    gemm(xh, d, Lw.wqkvh, d, 0, qkvf, 3L * d, rows, 3 * d, d,
         nullptr, nullptr, 1.f, 0, 1, 1, 0, 0, 0, 0, 0, 0);
    // QK LayerNorm over head dim (skip v chunks: row%24 >= 16)
    k_layernorm<<<rows * 24, 256, 0, stream>>>(qkvf, qkvf, nullptr, dh, 24, 16);
    pack(qkvf, qkvh, (long)rows * 3 * d);
    // scores = (Q K^T) * dim^-0.5   (batched over z = b*8 + h)
    float alpha = 1.f / sqrtf((float)d);
    gemm(qkvh, 3L * d, qkvh + d, 3L * d, 0, scoresf, n, n, n, dh,
         nullptr, nullptr, alpha, 0, 16, 8,
         (long)n * 3 * d, dh, (long)n * 3 * d, dh, 8L * n * n, (long)n * n);
    k_softmax<<<16 * n, 256, 0, stream>>>(scoresf, n);
    pack(scoresf, scoresh, 16L * n * n);
    // av = attn @ V   (B consumed as [K=n, N=dh] -> bLayout 1)
    gemm(scoresh, n, qkvh + 2 * d, 3L * d, 1, avf, d, n, dh, n,
         nullptr, nullptr, 1.f, 0, 16, 8,
         8L * n * n, (long)n * n, (long)n * 3 * d, dh, (long)n * d, dh);
    // parallel MLP: h1 = gelu(x W1^T + b1); par = h1 W2^T + b2
    gemm(xh, d, Lw.w1h, d, 0, h1f, m, rows, m, d,
         Lw.b1, nullptr, 1.f, 1, 1, 1, 0, 0, 0, 0, 0, 0);
    pack(h1f, h1h, (long)rows * m);
    gemm(h1h, m, Lw.w2h, m, 0, parf, d, rows, d, m,
         Lw.b2, nullptr, 1.f, 0, 1, 1, 0, 0, 0, 0, 0, 0);
    // attn_out + par, then x = ln(attn_out+par) + x
    pack(avf, avh, (long)rows * d);
    gemm(avh, d, Lw.woh, d, 0, sumf, d, rows, d, d,
         Lw.bo, parf, 1.f, 0, 1, 1, 0, 0, 0, 0, 0, 0);
    k_layernorm<<<rows, 256, 0, stream>>>(sumf, xb, xb, d, 0, 0);
  };

  // ---------------- weight packing (every call; no caching) ----------------
  pack(emb_w, embw_h, 1024L * 768);
  pack(out_w, outw_h, 768L * 1024);
  for (int i = 0; i < 7; ++i) {
    long d = dims[i];
    pack(L[i].wqkv, L[i].wqkvh, 3 * d * d);
    pack(L[i].wo,   L[i].woh,   d * d);
    pack(L[i].w1,   L[i].w1h,   4 * d * d);
    pack(L[i].w2,   L[i].w2h,   4 * d * d);
  }

  // ---------------- patch embedding ----------------
  k_patchify<<<gblocks(2L * 256 * 768), 256, 0, stream>>>(xin, xh);
  gemm(xh, 768, embw_h, 768, 0, bufX0, 1024, 512, 1024, 768,
       emb_b, nullptr, 1.f, 0, 1, 1, 0, 0, 0, 0, 0, 0);

  auto dap = [&](const float* s, float* dts, int n, int Dnew, int df, int step) {
    k_down_add_pos<<<gblocks(2L * n * Dnew), 256, 0, stream>>>(s, dts, pos, n, Dnew, df, step);
  };

  // ---------------- encoders + peak ----------------
  dap(bufX0, bufX0, 256, 1024, 1, 1);
  run_layer(L[0], bufX0, 256);
  hipMemcpyAsync(feat0, bufX0, (size_t)512 * 1024 * 4, hipMemcpyDeviceToDevice, stream);

  dap(bufX0, bufX1, 256, 512, 2, 2);
  run_layer(L[1], bufX1, 256);
  hipMemcpyAsync(feat1, bufX1, (size_t)512 * 512 * 4, hipMemcpyDeviceToDevice, stream);

  dap(bufX1, bufX0, 256, 256, 2, 4);
  run_layer(L[2], bufX0, 256);
  hipMemcpyAsync(feat2, bufX0, (size_t)512 * 256 * 4, hipMemcpyDeviceToDevice, stream);

  dap(bufX0, bufX1, 256, 128, 2, 8);
  run_layer(L[3], bufX1, 256);

  // ---------------- decoders (concat skip on token axis) ----------------
  k_copy_rows<<<gblocks(2L * 256 * 256), 256, 0, stream>>>(feat2, bufX0, 256, 256,
                                                           256L * 256, 512L * 256, 0);
  k_up_rows<<<gblocks(2L * 256 * 128), 256, 0, stream>>>(bufX1, bufX0, 256, 128,
                                                         256L * 128, 512L * 256, 256);
  run_layer(L[4], bufX0, 512);

  k_copy_rows<<<gblocks(2L * 256 * 512), 256, 0, stream>>>(feat1, bufX1, 256, 512,
                                                           256L * 512, 768L * 512, 0);
  k_up_rows<<<gblocks(2L * 512 * 256), 256, 0, stream>>>(bufX0, bufX1, 512, 256,
                                                         512L * 256, 768L * 512, 256);
  run_layer(L[5], bufX1, 768);

  k_copy_rows<<<gblocks(2L * 256 * 1024), 256, 0, stream>>>(feat0, bufX0, 256, 1024,
                                                            256L * 1024, 1024L * 1024, 0);
  k_up_rows<<<gblocks(2L * 768 * 512), 256, 0, stream>>>(bufX1, bufX0, 768, 512,
                                                         768L * 512, 1024L * 1024, 256);
  run_layer(L[6], bufX0, 1024);

  // ---------------- output projection + unpatchify ----------------
  pack(bufX0, xh, 2048L * 1024);
  gemm(xh, 1024, outw_h, 1024, 0, qkvf /*tf*/, 768, 2048, 768, 1024,
       out_b, nullptr, 1.f, 0, 1, 1, 0, 0, 0, 0, 0, 0);
  k_unpatchify<<<gblocks(2L * 12 * 65536), 256, 0, stream>>>(qkvf, y12);

  // ---------------- CNN tail ----------------
  k_conv3x3<<<16384, 256, 0, stream>>>(y12, cw1, cb1, c64, 12, 64, 0);
  k_inorm<<<128, 256, 0, stream>>>(c64, c64, nullptr, 1);
  k_conv3x3<<<16384, 256, 0, stream>>>(c64, cw2, cb2, c128A, 64, 128, 0);
  k_inorm<<<256, 256, 0, stream>>>(c128A, c128A, nullptr, 1);
  // residual block
  k_conv3x3<<<16384, 256, 0, stream>>>(c128A, rw1, rb1, c128B, 128, 128, 0);
  k_inorm<<<256, 256, 0, stream>>>(c128B, c128B, nullptr, 1);
  k_conv3x3<<<16384, 256, 0, stream>>>(c128B, rw2, rb2, c128C, 128, 128, 0);
  k_inorm<<<256, 256, 0, stream>>>(c128C, c128A, c128A, 0);   // y = y + inorm(conv)
  // tail convs
  k_conv3x3<<<16384, 256, 0, stream>>>(c128A, cw3, cb3, c64, 128, 64, 0);
  k_inorm<<<128, 256, 0, stream>>>(c64, c64, nullptr, 1);
  k_conv3x3<<<1536, 256, 0, stream>>>(c64, cw4, cb4, (float*)d_out, 64, 3, 2 /*tanh*/);
}
